// NodeProcessorModule_87608742903952
// MI455X (gfx1250) — matrix operand order repaired and checked by reading, not compiled, not run on hardware
//
#include <hip/hip_runtime.h>

#define D_FEAT 64
#define H_FEAT 128
#define K_IN   128   // 2*D

typedef __bf16 bf16_t;
typedef __attribute__((ext_vector_type(16))) bf16_t v16bf;
typedef __attribute__((ext_vector_type(8)))  bf16_t v8bf;
typedef __attribute__((ext_vector_type(4)))  bf16_t v4bf;
typedef __attribute__((ext_vector_type(8)))  float  v8f;

union AF { v16bf v; v8bf h[2]; bf16_t e[16]; };
union CF { v8f   v; float  f[8];  };
union P4 { v4bf  v; bf16_t e[4];  };

// ---------------------------------------------------------------------------
// Kernel 1: zero the aggregation scratch buffer (N x 64 f32 in d_ws)
// ---------------------------------------------------------------------------
__global__ void zero_agg(float* __restrict__ agg, long long n4) {
    long long i = (long long)blockIdx.x * blockDim.x + threadIdx.x;
    if (i < n4) ((float4*)agg)[i] = make_float4(0.f, 0.f, 0.f, 0.f);
}

// ---------------------------------------------------------------------------
// Kernel 2: scatter-sum messages. 16 threads per edge, float4 chunk each.
// unsafeAtomicAdd -> native global_atomic_add_f32 (no CAS loop).
// ---------------------------------------------------------------------------
__global__ void scatter_sum(const float* __restrict__ node_attr,
                            const int* __restrict__ senders,
                            const int* __restrict__ receivers,
                            float* __restrict__ agg, long long nE) {
    long long t = (long long)blockIdx.x * blockDim.x + threadIdx.x;
    long long e = t >> 4;            // edge index
    int part    = (int)(t & 15) * 4; // feature chunk [part, part+4)
    if (e >= nE) return;
    long long s = senders[e];
    long long r = receivers[e];
    const float4 v = *(const float4*)(node_attr + s * D_FEAT + part);
    float* dst = agg + r * D_FEAT + part;
    unsafeAtomicAdd(dst + 0, v.x);
    unsafeAtomicAdd(dst + 1, v.y);
    unsafeAtomicAdd(dst + 2, v.z);
    unsafeAtomicAdd(dst + 3, v.w);
}

// ---------------------------------------------------------------------------
// Kernel 3: fused 2-layer MLP via v_wmma_f32_16x16x32_bf16.
//   h   = relu([x | agg] @ W1 + b1)   (K=128, H=128)
//   out = h @ W2 + b2                 (K=128, N=64)
// Weights staged into LDS *pre-swizzled into B-fragment layout* so each
// fragment load is a contiguous 32B per lane (ds_load_b128 x2, no scalars).
// ---------------------------------------------------------------------------
__global__ __launch_bounds__(256)
void gnn_mlp_wmma(const float* __restrict__ x,    // [N,64]
                  const float* __restrict__ agg,  // [N,64]
                  const float* __restrict__ W1,   // [128,128] row-major (k,n)
                  const float* __restrict__ b1,   // [128]
                  const float* __restrict__ W2,   // [128,64]  row-major (k,n)
                  const float* __restrict__ b2,   // [64]
                  float* __restrict__ out,        // [N,64]
                  int nNodes) {
    // B-fragment-major weight storage: frag f=(nt*4+kc), lane l, elem e:
    //   k = kc*32 + ((l>>4)<<4) + e ,  n = nt*16 + (l&15)
    __shared__ __align__(16) bf16_t sW1f[32 * 32 * 16];  // 32 frags, 32 KB
    __shared__ __align__(16) bf16_t sW2f[16 * 32 * 16];  // 16 frags, 16 KB
    __shared__ __align__(16) bf16_t sT[8][16 * K_IN];    // 8 waves x 4 KB

    const int tid = threadIdx.x;

    // ---- stage W1 swizzled (global reads coalesced over n; b64 LDS stores) --
    for (int s = tid; s < (K_IN / 4) * H_FEAT; s += 256) {      // 4096 quads
        int n  = s & (H_FEAT - 1);
        int k0 = (s >> 7) << 2;                                  // quad of k
        P4 p;
        p.e[0] = (bf16_t)W1[(k0 + 0) * H_FEAT + n];
        p.e[1] = (bf16_t)W1[(k0 + 1) * H_FEAT + n];
        p.e[2] = (bf16_t)W1[(k0 + 2) * H_FEAT + n];
        p.e[3] = (bf16_t)W1[(k0 + 3) * H_FEAT + n];
        int kc = k0 >> 5, kl = k0 & 31;
        int lane_d = (n & 15) + (kl & 16);
        int fr = ((n >> 4) << 2) + kc;
        *(v4bf*)(sW1f + ((fr * 32 + lane_d) << 4) + (kl & 15)) = p.v;
    }
    // ---- stage W2 swizzled ----
    for (int s = tid; s < (H_FEAT / 4) * D_FEAT; s += 256) {    // 2048 quads
        int n  = s & (D_FEAT - 1);
        int k0 = (s >> 6) << 2;
        P4 p;
        p.e[0] = (bf16_t)W2[(k0 + 0) * D_FEAT + n];
        p.e[1] = (bf16_t)W2[(k0 + 1) * D_FEAT + n];
        p.e[2] = (bf16_t)W2[(k0 + 2) * D_FEAT + n];
        p.e[3] = (bf16_t)W2[(k0 + 3) * D_FEAT + n];
        int kc = k0 >> 5, kl = k0 & 31;
        int lane_d = (n & 15) + (kl & 16);
        int fr = ((n >> 4) << 2) + kc;
        *(v4bf*)(sW2f + ((fr * 32 + lane_d) << 4) + (kl & 15)) = p.v;
    }
    __syncthreads();

    const int wave = tid >> 5;
    const int lane = tid & 31;
    const int base = (blockIdx.x * 8 + wave) * 16;  // first node row of tile
    if (base >= nNodes) return;                     // wave-uniform branch

    bf16_t* tile = sT[wave];

    // ---- stage 16x128 concat([x, agg]) tile: float4 load -> b64 bf16 store --
    for (int it = lane; it < (16 * K_IN) / 4; it += 32) {       // 512 quads
        int m  = it >> 5;
        int k0 = (it & 31) << 2;
        int row = base + m; if (row >= nNodes) row = nNodes - 1;
        const float4 v4 = (k0 < D_FEAT)
            ? *(const float4*)(x   + (long long)row * D_FEAT + k0)
            : *(const float4*)(agg + (long long)row * D_FEAT + (k0 - D_FEAT));
        P4 p;
        p.e[0] = (bf16_t)v4.x; p.e[1] = (bf16_t)v4.y;
        p.e[2] = (bf16_t)v4.z; p.e[3] = (bf16_t)v4.w;
        *(v4bf*)(tile + m * K_IN + k0) = p.v;
    }
    // per-wave LDS RAW: DS ops are in-order within a wave.

    // 16-bit fragment coordinates as used here (A and B share contiguous k = 16*half + e):
    const int mA  = lane & 15;          // A: lane -> row M
    const int hiA = (lane >> 4) << 4;   // A: upper half-lanes hold K+16
    const int nB  = lane & 15;          // B/C: lane -> col N
    const int mC  = (lane >> 4) << 3;   // C/D: upper half-lanes hold M+8

    // A fragment: two contiguous 8-elem runs -> 2x 16B LDS loads
    auto loadA = [&](const bf16_t* t, int kc) {
        AF a;
        const bf16_t* p = t + mA * K_IN + kc * 32 + hiA;
        a.h[0] = *(const v8bf*)(p);
        a.h[1] = *(const v8bf*)(p + 8);
        return a;
    };
    // B fragment: contiguous 32B per lane from swizzled weight store
    auto loadB = [&](const bf16_t* w, int nt, int kc) {
        AF b;
        b.v = *(const v16bf*)(w + ((((nt << 2) + kc) * 32 + lane) << 4));
        return b;
    };

    // ---- GEMM1: h = relu([x|agg] @ W1 + b1) ----
    AF a1[4];
#pragma unroll
    for (int kc = 0; kc < 4; ++kc) a1[kc] = loadA(tile, kc);

    CF acc[8];
#pragma unroll
    for (int nt = 0; nt < 8; ++nt) {
        float bias = b1[nt * 16 + nB];
#pragma unroll
        for (int r = 0; r < 8; ++r) acc[nt].f[r] = bias;
#pragma unroll
        for (int kc = 0; kc < 4; ++kc) {
            AF b = loadB(sW1f, nt, kc);
            acc[nt].v = __builtin_amdgcn_wmma_f32_16x16x32_bf16(
                false, a1[kc].v, false, b.v, (short)0, acc[nt].v, false, false);
        }
    }

    // ---- ReLU, write h (bf16) back over the per-wave tile ----
#pragma unroll
    for (int nt = 0; nt < 8; ++nt)
#pragma unroll
        for (int r = 0; r < 8; ++r) {
            float v = acc[nt].f[r];
            v = v > 0.f ? v : 0.f;
            tile[(r + mC) * K_IN + nt * 16 + nB] = (bf16_t)v;
        }

    // ---- GEMM2: out = h @ W2 + b2 ----
    AF a2[4];
#pragma unroll
    for (int kc = 0; kc < 4; ++kc) a2[kc] = loadA(tile, kc);

#pragma unroll
    for (int nt = 0; nt < 4; ++nt) {
        CF o;
        float bias = b2[nt * 16 + nB];
#pragma unroll
        for (int r = 0; r < 8; ++r) o.f[r] = bias;
#pragma unroll
        for (int kc = 0; kc < 4; ++kc) {
            AF b = loadB(sW2f, nt, kc);
            o.v = __builtin_amdgcn_wmma_f32_16x16x32_bf16(
                false, a2[kc].v, false, b.v, (short)0, o.v, false, false);
        }
#pragma unroll
        for (int r = 0; r < 8; ++r) {
            int row = base + r + mC;
            if (row < nNodes)
                out[(long long)row * D_FEAT + nt * 16 + nB] = o.f[r];
        }
    }
}

// ---------------------------------------------------------------------------
extern "C" void kernel_launch(void* const* d_in, const int* in_sizes, int n_in,
                              void* d_out, int out_size, void* d_ws, size_t ws_size,
                              hipStream_t stream) {
    const float* node_attr = (const float*)d_in[0];
    const int*   edge      = (const int*)d_in[1];   // [2, E] int
    const float* W1 = (const float*)d_in[2];
    const float* b1 = (const float*)d_in[3];
    const float* W2 = (const float*)d_in[4];
    const float* b2 = (const float*)d_in[5];
    float* out = (float*)d_out;
    float* agg = (float*)d_ws;                      // N x 64 f32 scratch

    const int nNodes   = in_sizes[0] / D_FEAT;
    const long long nE = in_sizes[1] / 2;
    const int* senders   = edge;
    const int* receivers = edge + nE;

    // 1) zero agg
    long long n4 = (long long)nNodes * D_FEAT / 4;
    int zb = (int)((n4 + 255) / 256);
    zero_agg<<<zb, 256, 0, stream>>>(agg, n4);

    // 2) scatter-sum (16 threads per edge)
    long long st = nE * 16;
    int sb = (int)((st + 255) / 256);
    scatter_sum<<<sb, 256, 0, stream>>>(node_attr, senders, receivers, agg, nE);

    // 3) fused WMMA MLP (8 tiles of 16 nodes per 256-thread block)
    int mb = (nNodes + 127) / 128;
    gnn_mlp_wmma<<<mb, 256, 0, stream>>>(node_attr, agg, W1, b1, W2, b2, out, nNodes);
}
